// BioHGTLayer_38259568673212
// MI455X (gfx1250) — compile-verified, matrix operand order checked by reading, not run on hardware
//
#include <hip/hip_runtime.h>
#include <math.h>

// ---------------- problem constants (from reference) ----------------
#define TT 4
#define NN 50000
#define DD 256
#define HH 8
#define RR 4
#define EE 200000
#define DKK 32

typedef __attribute__((ext_vector_type(16))) __bf16 v16bf;
typedef __attribute__((ext_vector_type(8)))  __bf16 v8bf;
typedef __attribute__((ext_vector_type(8)))  float  v8f;

// =====================================================================
// WMMA bf16 GEMM:  out[t] = A[t] @ W[t] + bias[t]   (optionally GELU)
// A: [T, M, K] fp32, W: [T, K, Nout] fp32, bias: [T, Nout]
// block = 256 threads = 8 waves; block tile 64(M) x 64(N); K-step 32
// wave (wm,wn) in 4x2 grid computes a 16x32 strip = 2 WMMA tiles / K-step
// LDS: A row-major bf16 (stride 40), W transposed bf16 (stride 40) so every
// fragment operand is two contiguous 16-byte LDS vectors (ds_load_b128).
// =====================================================================
#define BM 64
#define BN 64
#define BKs 32
#define LSTR 40   // padded LDS row stride (elements): 80B rows, conflict-free

template <int EPI> // 0 = bias, 1 = bias + exact GELU
__global__ __launch_bounds__(256) void gemm_wmma(
    const float* __restrict__ A, const float* __restrict__ W,
    const float* __restrict__ bias, float* __restrict__ out,
    int M, int K, int Nout)
{
    const int t = blockIdx.z;
    const float* At = A    + (size_t)t * M * K;
    const float* Wt = W    + (size_t)t * K * Nout;
    const float* bt = bias + (size_t)t * Nout;
    float*       Ot = out  + (size_t)t * M * Nout;

    __shared__ __bf16 Ash[BM * LSTR];     // [row][k]   64x40
    __shared__ __bf16 WshT[BN * LSTR];    // [col][k]   64x40 (transposed W)

    const int tid  = threadIdx.x;
    const int lane = tid & 31;
    const int wave = tid >> 5;      // 0..7
    const int wm   = wave & 3;      // 0..3 : M sub-tile (16 rows)
    const int wn   = wave >> 2;     // 0..1 : N sub-tile (32 cols)

    const int m0 = blockIdx.y * BM;
    const int n0 = blockIdx.x * BN;

    // staging coords (branch-free, vectorized)
    const int ar = tid >> 2;            // A row 0..63
    const int ac = (tid & 3) * 8;       // A col start {0,8,16,24}
    int agr = m0 + ar; if (agr >= M) agr = M - 1;      // clamp: only garbage
    const size_t aBase = (size_t)agr * K + ac;         // rows feed garbage outs

    const int wr = tid >> 3;            // W k-row 0..31
    const int wc = (tid & 7) * 8;       // W col start {0..56}

    // fragment addresses (per ISA 16-bit A/B layout)
    const int mrow  = (lane & 15) + wm * 16;
    const int ncol  = (lane & 15) + wn * 32;
    const int khalf = (lane >> 4) * 8;
    const __bf16* aptr  = &Ash [mrow * LSTR + khalf];
    const __bf16* bptr0 = &WshT[ ncol        * LSTR + khalf];
    const __bf16* bptr1 = &WshT[(ncol + 16)  * LSTR + khalf];

    v8f acc0 = {}, acc1 = {};

    for (int k0 = 0; k0 < K; k0 += BKs) {
        // ---- stage A: 2 x global_load_b128, 1 x ds_store_b128 ----
        {
            const float4* ap = reinterpret_cast<const float4*>(At + aBase + k0);
            float4 f0 = ap[0], f1 = ap[1];
            v8bf pk = { (__bf16)f0.x, (__bf16)f0.y, (__bf16)f0.z, (__bf16)f0.w,
                        (__bf16)f1.x, (__bf16)f1.y, (__bf16)f1.z, (__bf16)f1.w };
            *reinterpret_cast<v8bf*>(&Ash[ar * LSTR + ac]) = pk;
        }
        // ---- stage W transposed: 2 x global_load_b128, 8 scalar b16 LDS ----
        {
            const float4* wp = reinterpret_cast<const float4*>(
                Wt + (size_t)(k0 + wr) * Nout + n0 + wc);
            float4 g0 = wp[0], g1 = wp[1];
            WshT[(wc + 0) * LSTR + wr] = (__bf16)g0.x;
            WshT[(wc + 1) * LSTR + wr] = (__bf16)g0.y;
            WshT[(wc + 2) * LSTR + wr] = (__bf16)g0.z;
            WshT[(wc + 3) * LSTR + wr] = (__bf16)g0.w;
            WshT[(wc + 4) * LSTR + wr] = (__bf16)g1.x;
            WshT[(wc + 5) * LSTR + wr] = (__bf16)g1.y;
            WshT[(wc + 6) * LSTR + wr] = (__bf16)g1.z;
            WshT[(wc + 7) * LSTR + wr] = (__bf16)g1.w;
        }
        __syncthreads();

        // ---- fragments: each operand = two contiguous 16B LDS vectors ----
        v8bf alo = *reinterpret_cast<const v8bf*>(aptr);
        v8bf ahi = *reinterpret_cast<const v8bf*>(aptr + 16);
        v8bf b0l = *reinterpret_cast<const v8bf*>(bptr0);
        v8bf b0h = *reinterpret_cast<const v8bf*>(bptr0 + 16);
        v8bf b1l = *reinterpret_cast<const v8bf*>(bptr1);
        v8bf b1h = *reinterpret_cast<const v8bf*>(bptr1 + 16);
        v16bf a  = __builtin_shufflevector(alo, ahi, 0,1,2,3,4,5,6,7,8,9,10,11,12,13,14,15);
        v16bf b0 = __builtin_shufflevector(b0l, b0h, 0,1,2,3,4,5,6,7,8,9,10,11,12,13,14,15);
        v16bf b1 = __builtin_shufflevector(b1l, b1h, 0,1,2,3,4,5,6,7,8,9,10,11,12,13,14,15);

        acc0 = __builtin_amdgcn_wmma_f32_16x16x32_bf16(
                   false, a, false, b0, (short)0, acc0, false, false);
        acc1 = __builtin_amdgcn_wmma_f32_16x16x32_bf16(
                   false, a, false, b1, (short)0, acc1, false, false);
        __syncthreads();
    }

    // ---- epilogue: C/D layout VGPR r -> M = r + 8*(lane>=16), N = lane&15
    const int col0 = n0 + wn * 32 + (lane & 15);
    const int col1 = col0 + 16;
    const float bv0 = bt[col0];
    const float bv1 = bt[col1];
    const int rbase = m0 + wm * 16 + ((lane >> 4) * 8);

    if (m0 + BM <= M) {                 // fast path: full tile, no guards
        #pragma unroll
        for (int r = 0; r < 8; ++r) {
            float v0 = acc0[r] + bv0;
            float v1 = acc1[r] + bv1;
            if (EPI == 1) {
                v0 = 0.5f * v0 * (1.0f + erff(v0 * 0.70710678118654752f));
                v1 = 0.5f * v1 * (1.0f + erff(v1 * 0.70710678118654752f));
            }
            Ot[(size_t)(rbase + r) * Nout + col0] = v0;
            Ot[(size_t)(rbase + r) * Nout + col1] = v1;
        }
    } else {
        #pragma unroll
        for (int r = 0; r < 8; ++r) {
            int row = rbase + r;
            if (row < M) {
                float v0 = acc0[r] + bv0;
                float v1 = acc1[r] + bv1;
                if (EPI == 1) {
                    v0 = 0.5f * v0 * (1.0f + erff(v0 * 0.70710678118654752f));
                    v1 = 0.5f * v1 * (1.0f + erff(v1 * 0.70710678118654752f));
                }
                Ot[(size_t)row * Nout + col0] = v0;
                Ot[(size_t)row * Nout + col1] = v1;
            }
        }
    }
}

// =====================================================================
// monotone float<->uint mapping for atomicMax-based segment max
// =====================================================================
__device__ __forceinline__ unsigned fenc(float f) {
    unsigned u = __float_as_uint(f);
    return (u & 0x80000000u) ? ~u : (u | 0x80000000u);
}
__device__ __forceinline__ float fdec(unsigned k) {
    unsigned u = (k & 0x80000000u) ? (k & 0x7FFFFFFFu) : ~k;
    return __uint_as_float(u);
}

__device__ __forceinline__ void load32(const float* __restrict__ p, float* v) {
    const float4* p4 = reinterpret_cast<const float4*>(p);
    #pragma unroll
    for (int i = 0; i < 8; ++i) {
        float4 tq = p4[i];
        v[4*i+0] = tq.x; v[4*i+1] = tq.y; v[4*i+2] = tq.z; v[4*i+3] = tq.w;
    }
}

// ---------------- edge pass 1: logits + segment max -------------------
__global__ __launch_bounds__(256) void edge_logits(
    const float* __restrict__ Qdt, const float* __restrict__ Kst,
    const int* __restrict__ src, const int* __restrict__ dst,
    const float* __restrict__ Wattn_r,          // [H][DK][DK]
    const float* __restrict__ rel_bias, int r,
    float* __restrict__ logits, unsigned* __restrict__ maxbuf)
{
    __shared__ float Ws[HH * DKK * DKK];        // 32 KB
    for (int i = threadIdx.x; i < HH * DKK * DKK; i += 256) Ws[i] = Wattn_r[i];
    __syncthreads();

    const int gid = blockIdx.x * 256 + threadIdx.x;
    if (gid >= EE * HH) return;
    const int e = gid >> 3;
    const int h = gid & 7;
    const int s = src[e], d = dst[e];

    float qv[DKK], kv[DKK];
    load32(Qdt + (size_t)d * DD + h * DKK, qv);
    load32(Kst + (size_t)s * DD + h * DKK, kv);

    const float* W = Ws + h * DKK * DKK;
    float logit = 0.0f;
    #pragma unroll
    for (int j = 0; j < DKK; ++j) {
        float kr = 0.0f;
        #pragma unroll
        for (int i = 0; i < DKK; ++i) kr += kv[i] * W[i * DKK + j];
        logit += qv[j] * kr;
    }
    logit = logit * 0.17677669529663687f + rel_bias[r];   // 1/sqrt(32)

    logits[gid] = logit;
    atomicMax(&maxbuf[(size_t)d * HH + h], fenc(logit));
}

// ---------------- edge pass 2: exp + segment sum ----------------------
__global__ __launch_bounds__(256) void edge_exp(
    const int* __restrict__ dst,
    float* __restrict__ logits,                  // in: logit, out: p
    const unsigned* __restrict__ maxbuf, float* __restrict__ zbuf)
{
    const int gid = blockIdx.x * 256 + threadIdx.x;
    if (gid >= EE * HH) return;
    const int e = gid >> 3;
    const int h = gid & 7;
    const int d = dst[e];
    float m = fdec(maxbuf[(size_t)d * HH + h]);
    float p = expf(logits[gid] - m);
    logits[gid] = p;
    atomicAdd(&zbuf[(size_t)d * HH + h], p);
}

// ---------------- edge pass 3: message + scatter-add ------------------
__global__ __launch_bounds__(256) void edge_scatter(
    const float* __restrict__ Vst,
    const int* __restrict__ src, const int* __restrict__ dst,
    const float* __restrict__ Wmsg_r,
    const float* __restrict__ pbuf, const float* __restrict__ zbuf,
    float* __restrict__ agg_dt)
{
    __shared__ float Ws[HH * DKK * DKK];
    for (int i = threadIdx.x; i < HH * DKK * DKK; i += 256) Ws[i] = Wmsg_r[i];
    __syncthreads();

    const int gid = blockIdx.x * 256 + threadIdx.x;
    if (gid >= EE * HH) return;
    const int e = gid >> 3;
    const int h = gid & 7;
    const int s = src[e], d = dst[e];

    const float attn = pbuf[gid] / zbuf[(size_t)d * HH + h];

    float vv[DKK];
    load32(Vst + (size_t)s * DD + h * DKK, vv);

    const float* W = Ws + h * DKK * DKK;
    float* out = agg_dt + (size_t)d * DD + h * DKK;
    #pragma unroll
    for (int j = 0; j < DKK; ++j) {
        float vr = 0.0f;
        #pragma unroll
        for (int i = 0; i < DKK; ++i) vr += vv[i] * W[i * DKK + j];
        atomicAdd(&out[j], attn * vr);
    }
}

// ---------------- layernorm: out = LN(a+b)*g + be  (one wave32 / row) -
__global__ __launch_bounds__(256) void ln_kernel(
    const float* __restrict__ a, const float* __restrict__ b,
    const float* __restrict__ g, const float* __restrict__ be,
    float* __restrict__ out, int rowsTotal)
{
    const int wave = threadIdx.x >> 5;
    const int lane = threadIdx.x & 31;
    const int row  = blockIdx.x * 8 + wave;
    if (row >= rowsTotal) return;
    const int t = row / NN;

    const float4* a4 = reinterpret_cast<const float4*>(a + (size_t)row * DD) + lane * 2;
    const float4* b4 = reinterpret_cast<const float4*>(b + (size_t)row * DD) + lane * 2;
    float v[8];
    {
        float4 x0 = a4[0], x1 = a4[1], y0 = b4[0], y1 = b4[1];
        v[0] = x0.x + y0.x; v[1] = x0.y + y0.y; v[2] = x0.z + y0.z; v[3] = x0.w + y0.w;
        v[4] = x1.x + y1.x; v[5] = x1.y + y1.y; v[6] = x1.z + y1.z; v[7] = x1.w + y1.w;
    }
    float s = 0.0f;
    #pragma unroll
    for (int i = 0; i < 8; ++i) s += v[i];
    #pragma unroll
    for (int off = 16; off > 0; off >>= 1) s += __shfl_xor(s, off, 32);
    const float mu = s * (1.0f / DD);

    float vs = 0.0f;
    #pragma unroll
    for (int i = 0; i < 8; ++i) { float dv = v[i] - mu; vs += dv * dv; }
    #pragma unroll
    for (int off = 16; off > 0; off >>= 1) vs += __shfl_xor(vs, off, 32);
    const float rstd = rsqrtf(vs * (1.0f / DD) + 1e-5f);

    const float4* g4  = reinterpret_cast<const float4*>(g  + (size_t)t * DD) + lane * 2;
    const float4* be4 = reinterpret_cast<const float4*>(be + (size_t)t * DD) + lane * 2;
    float4* o4 = reinterpret_cast<float4*>(out + (size_t)row * DD) + lane * 2;
    float4 gg0 = g4[0], gg1 = g4[1], bb0 = be4[0], bb1 = be4[1];
    float4 r0, r1;
    r0.x = (v[0]-mu)*rstd*gg0.x + bb0.x; r0.y = (v[1]-mu)*rstd*gg0.y + bb0.y;
    r0.z = (v[2]-mu)*rstd*gg0.z + bb0.z; r0.w = (v[3]-mu)*rstd*gg0.w + bb0.w;
    r1.x = (v[4]-mu)*rstd*gg1.x + bb1.x; r1.y = (v[5]-mu)*rstd*gg1.y + bb1.y;
    r1.z = (v[6]-mu)*rstd*gg1.z + bb1.z; r1.w = (v[7]-mu)*rstd*gg1.w + bb1.w;
    o4[0] = r0; o4[1] = r1;
}

// ---------------- fills ----------------------------------------------
__global__ void fill_f32(float* p, float v, size_t n) {
    size_t i = (size_t)blockIdx.x * 256 + threadIdx.x;
    if (i < n) p[i] = v;
}
__global__ void fill_u32(unsigned* p, unsigned v, size_t n) {
    size_t i = (size_t)blockIdx.x * 256 + threadIdx.x;
    if (i < n) p[i] = v;
}

// =====================================================================
extern "C" void kernel_launch(void* const* d_in, const int* in_sizes, int n_in,
                              void* d_out, int out_size, void* d_ws, size_t ws_size,
                              hipStream_t stream)
{
    (void)in_sizes; (void)n_in; (void)out_size; (void)ws_size;
    const float* x        = (const float*)d_in[0];
    const int*   edge_src = (const int*)  d_in[1];
    const int*   edge_dst = (const int*)  d_in[2];
    const float* Wq  = (const float*)d_in[3];
    const float* bq  = (const float*)d_in[4];
    const float* Wk  = (const float*)d_in[5];
    const float* bk  = (const float*)d_in[6];
    const float* Wv  = (const float*)d_in[7];
    const float* bv  = (const float*)d_in[8];
    const float* W_attn   = (const float*)d_in[9];
    const float* W_msg    = (const float*)d_in[10];
    const float* rel_bias = (const float*)d_in[11];
    const float* ln1_g = (const float*)d_in[12];
    const float* ln1_b = (const float*)d_in[13];
    const float* ln2_g = (const float*)d_in[14];
    const float* ln2_b = (const float*)d_in[15];
    const float* ffn_w1 = (const float*)d_in[16];
    const float* ffn_b1 = (const float*)d_in[17];
    const float* ffn_w2 = (const float*)d_in[18];
    const float* ffn_b2 = (const float*)d_in[19];

    // ---- workspace carve-up ----
    char* ws = (char*)d_ws;
    size_t off = 0;
    auto alloc = [&](size_t bytes) -> void* {
        void* p = ws + off;
        off += (bytes + 255) & ~(size_t)255;
        return p;
    };
    const size_t szND = (size_t)TT * NN * DD;          // 51.2M elems
    float*    Qn   = (float*)   alloc(szND * 4);
    float*    Kn   = (float*)   alloc(szND * 4);
    float*    Vn   = (float*)   alloc(szND * 4);
    float*    agg  = (float*)   alloc(szND * 4);       // reused as FFN2 output
    float*    hbuf = (float*)   alloc(szND * 4);
    float*    fbuf = (float*)   alloc((size_t)TT * NN * 4 * DD * 4);
    float*    logits = (float*) alloc((size_t)EE * HH * 4);
    unsigned* maxb   = (unsigned*)alloc((size_t)NN * HH * 4);
    float*    zb     = (float*) alloc((size_t)NN * HH * 4);

    const dim3 blk(256);
    const int mt = (NN + BM - 1) / BM;                 // 782 M-tiles

    // ---- Q/K/V projections (WMMA GEMMs) ----
    {
        dim3 grid(DD / BN, mt, TT);
        gemm_wmma<0><<<grid, blk, 0, stream>>>(x, Wq, bq, Qn, NN, DD, DD);
        gemm_wmma<0><<<grid, blk, 0, stream>>>(x, Wk, bk, Kn, NN, DD, DD);
        gemm_wmma<0><<<grid, blk, 0, stream>>>(x, Wv, bv, Vn, NN, DD, DD);
    }

    // ---- zero agg ----
    fill_f32<<<(unsigned)((szND + 255) / 256), blk, 0, stream>>>(agg, 0.0f, szND);

    // ---- relation passes ----
    const int st_tab[RR] = {1, 2, 3, 1};
    const int dt_tab[RR] = {0, 0, 3, 3};
    const size_t nh = (size_t)NN * HH;
    const unsigned ehBlocks = (unsigned)(((size_t)EE * HH + 255) / 256);
    for (int r = 0; r < RR; ++r) {
        const int st = st_tab[r], dt = dt_tab[r];
        fill_u32<<<(unsigned)((nh + 255) / 256), blk, 0, stream>>>(maxb, 0u, nh);
        fill_f32<<<(unsigned)((nh + 255) / 256), blk, 0, stream>>>(zb, 0.0f, nh);

        const int* src = edge_src + (size_t)r * EE;
        const int* dst = edge_dst + (size_t)r * EE;

        edge_logits<<<ehBlocks, blk, 0, stream>>>(
            Qn + (size_t)dt * NN * DD, Kn + (size_t)st * NN * DD,
            src, dst, W_attn + (size_t)r * HH * DKK * DKK,
            rel_bias, r, logits, maxb);

        edge_exp<<<ehBlocks, blk, 0, stream>>>(dst, logits, maxb, zb);

        edge_scatter<<<ehBlocks, blk, 0, stream>>>(
            Vn + (size_t)st * NN * DD, src, dst,
            W_msg + (size_t)r * HH * DKK * DKK,
            logits, zb, agg + (size_t)dt * NN * DD);
    }

    // ---- h = LN1(x + agg) ----
    const int rowsTotal = TT * NN;
    ln_kernel<<<(rowsTotal + 7) / 8, blk, 0, stream>>>(x, agg, ln1_g, ln1_b, hbuf, rowsTotal);

    // ---- FFN (WMMA GEMMs): f = gelu(h@W1+b1); f2 = f@W2+b2 (into agg) ----
    {
        dim3 g1(4 * DD / BN, mt, TT);
        gemm_wmma<1><<<g1, blk, 0, stream>>>(hbuf, ffn_w1, ffn_b1, fbuf, NN, DD, 4 * DD);
        dim3 g2(DD / BN, mt, TT);
        gemm_wmma<0><<<g2, blk, 0, stream>>>(fbuf, ffn_w2, ffn_b2, agg, NN, 4 * DD, DD);
    }

    // ---- out = LN2(h + f2) ----
    ln_kernel<<<(rowsTotal + 7) / 8, blk, 0, stream>>>(hbuf, agg, ln2_g, ln2_b, (float*)d_out, rowsTotal);
}
